// Model_24163486007580
// MI455X (gfx1250) — matrix operand order repaired and checked by reading, not compiled, not run on hardware
//
#include <hip/hip_runtime.h>
#include <math.h>

#define BRAYS 1024
#define NSC   64
#define HDIM  256
#define SAMPLE_DIST (2.0f / 64.0f)

typedef __attribute__((ext_vector_type(16))) _Float16 v16h;
typedef __attribute__((ext_vector_type(8)))  float    v8f;

union Frag16 { v16h h; unsigned int u[8]; };

__device__ __forceinline__ float sigmoidf_dev(float x) { return 1.f / (1.f + __expf(-x)); }

// ---------------------------------------------------------------------------
// WMMA fragment loaders (v_wmma_f32_16x16x32_f16 layouts per CDNA5 ISA 7.12.2)
// ---------------------------------------------------------------------------
// A: 16x32 f16. lane m=L&15 (row), half=L>>4. VGPR v: K = (v>>2)*16 + half*8 + (v&3)*2
__device__ __forceinline__ v16h load_a_frag(const _Float16* X, int stride, int kBase, int lane) {
  Frag16 f;
  const int m = lane & 15, half = lane >> 4;
  const _Float16* row = X + m * stride + kBase + (half << 3);
#pragma unroll
  for (int v = 0; v < 8; ++v) {
    const int k = ((v >> 2) << 4) + ((v & 3) << 1);
    f.u[v] = *(const unsigned int*)(row + k);
  }
  return f.h;
}

// B: 32x16 f16 from transposed weights Wt[n][k] (row n contiguous over k).
// lane n=L&15 (col), half=L>>4. VGPR v: K = (v>>2)*16 + half*8 + (v&3)*2
__device__ __forceinline__ v16h load_b_frag(const _Float16* Wt, int K, int kBase, int nBase, int lane) {
  Frag16 f;
  const int n = nBase + (lane & 15), half = lane >> 4;
  const _Float16* row = Wt + n * K + kBase + (half << 3);
#pragma unroll
  for (int v = 0; v < 8; ++v) f.u[v] = *(const unsigned int*)(row + (((v >> 2) << 4) + ((v & 3) << 1)));
  return f.h;
}

// ---------------------------------------------------------------------------
// Tile epilogue: bias + activation (+relu-mask) + store, fully specialized.
// C layout: lane holds col nc = nBase+(L&15), rows v+(L>>4)*8, v=0..7.
// ---------------------------------------------------------------------------
template <int NOUTPAD, int ACT, bool HAS_BIAS, bool HAS_MASK, bool OUT_H, bool OUT_F>
__device__ __forceinline__ void store_tile(v8f c, int nBase, const float* __restrict__ bias,
                                           _Float16* Yh, int yStride, float* Yf,
                                           const _Float16* maskBuf, int maskStride, int lane)
{
  const int nc = nBase + (lane & 15);
  const int half = lane >> 4;
  const float bv = HAS_BIAS ? bias[nc] : 0.f;
#pragma unroll
  for (int v = 0; v < 8; ++v) {
    const int row = v + (half << 3);
    float y = c[v] + bv;
    if constexpr (ACT == 1) y = fmaxf(y, 0.f);
    if constexpr (HAS_MASK) {
      if (!((float)maskBuf[row * maskStride + nc] > 0.f)) y = 0.f;
    }
    if constexpr (OUT_H) Yh[row * yStride + nc] = (_Float16)y;
    if constexpr (OUT_F) Yf[row * NOUTPAD + nc] = y;
  }
}

// ---------------------------------------------------------------------------
// Generic LDS-resident MLP layer for a 16-point tile, 256-thread block (8 waves).
// Y = act(X @ Wt^T + bias). Compile-time specialized (no runtime branches in
// the hot loop). For NOUTPAD==256 each wave owns two 16-col tiles: one shared
// A-fragment feeds two back-to-back WMMAs (dual accumulators amortize the
// D->VALU hazard). K is a template constant so the K-loop fully unrolls.
// ---------------------------------------------------------------------------
template <int K, int NOUTPAD, int ACT, bool HAS_BIAS, bool HAS_MASK, bool OUT_H, bool OUT_F>
__device__ void mlp_layer(const _Float16* __restrict__ X, int xStride,
                          const _Float16* __restrict__ Wt, const float* __restrict__ bias,
                          _Float16* Yh, int yStride, float* Yf,
                          const _Float16* maskBuf, int maskStride)
{
  const int lane = threadIdx.x & 31;
  const int wave = threadIdx.x >> 5;
  if constexpr (NOUTPAD == 256) {
    const int n0 = wave << 4;
    const int n1 = (wave + 8) << 4;
    __builtin_prefetch(Wt + (size_t)n0 * K, 0, 1);   // global_prefetch_b8
    __builtin_prefetch(Wt + (size_t)n1 * K, 0, 1);
    v8f c0 = {}, c1 = {};
#pragma unroll
    for (int kb = 0; kb < K; kb += 32) {
      v16h a  = load_a_frag(X, xStride, kb, lane);
      v16h b0 = load_b_frag(Wt, K, kb, n0, lane);
      v16h b1 = load_b_frag(Wt, K, kb, n1, lane);
      c0 = __builtin_amdgcn_wmma_f32_16x16x32_f16(false, a, false, b0, (short)0, c0, false, false);
      c1 = __builtin_amdgcn_wmma_f32_16x16x32_f16(false, a, false, b1, (short)0, c1, false, false);
    }
    store_tile<NOUTPAD, ACT, HAS_BIAS, HAS_MASK, OUT_H, OUT_F>(
        c0, n0, bias, Yh, yStride, Yf, maskBuf, maskStride, lane);
    store_tile<NOUTPAD, ACT, HAS_BIAS, HAS_MASK, OUT_H, OUT_F>(
        c1, n1, bias, Yh, yStride, Yf, maskBuf, maskStride, lane);
  } else {
    // single 16-col tile, computed by wave 0 (wave-uniform branch; EXEC intact)
    if (wave == 0) {
      v8f c = {};
#pragma unroll
      for (int kb = 0; kb < K; kb += 32) {
        v16h a = load_a_frag(X, xStride, kb, lane);
        v16h b = load_b_frag(Wt, K, kb, 0, lane);
        c = __builtin_amdgcn_wmma_f32_16x16x32_f16(false, a, false, b, (short)0, c, false, false);
      }
      store_tile<NOUTPAD, ACT, HAS_BIAS, HAS_MASK, OUT_H, OUT_F>(
          c, 0, bias, Yh, yStride, Yf, maskBuf, maskStride, lane);
    }
  }
}

// ---------------------------------------------------------------------------
// Weight prep kernels (run every call; deterministic)
// ---------------------------------------------------------------------------
// dst[n][k] = (k<K && n<N) ? src[k*srcStride + n] : 0   (f32 -> f16 transpose+pad)
__global__ void w_transpose_pad(_Float16* dst, const float* src,
                                int K, int N, int srcStride, int KP, int NP)
{
  int i = blockIdx.x * blockDim.x + threadIdx.x;
  if (i >= KP * NP) return;
  int n = i / KP, k = i - n * KP;
  float v = (k < K && n < N) ? src[k * srcStride + n] : 0.f;
  dst[n * KP + k] = (_Float16)v;
}

// dst[r][c] = (r<R && c<C) ? src[r*C + c] : 0   (f32 -> f16 pad, no transpose)
__global__ void w_copy_pad(_Float16* dst, const float* src, int R, int C, int RP, int CP)
{
  int i = blockIdx.x * blockDim.x + threadIdx.x;
  if (i >= RP * CP) return;
  int r = i / CP, c = i - r * CP;
  float v = (r < R && c < C) ? src[r * C + c] : 0.f;
  dst[r * CP + c] = (_Float16)v;
}

__global__ void bias_pad(float* dst, const float* src, int n, int np)
{
  int i = blockIdx.x * blockDim.x + threadIdx.x;
  if (i >= np) return;
  dst[i] = (i < n) ? src[i] : 0.f;
}

// ---------------------------------------------------------------------------
// z initialization: z[r][s] = base + span * s/(S-1); base/span from near/far or scalars
// ---------------------------------------------------------------------------
__global__ void init_z_kernel(float* z, int S, const float* nearp, const float* farp,
                              float base, float span)
{
  int i = blockIdx.x * blockDim.x + threadIdx.x;
  if (i >= BRAYS * S) return;
  int r = i / S, s = i - r * S;
  float b = nearp ? nearp[r] : base;
  float sp = farp ? (farp[r] - nearp[r]) : span;
  z[i] = b + sp * ((float)s / (float)(S - 1));
}

// ---------------------------------------------------------------------------
// SDF value kernel (importance passes): fused 3->256->256->1 MLP, WMMA
// ---------------------------------------------------------------------------
__global__ void __launch_bounds__(256) sdf_value_kernel(
    const float* __restrict__ o, const float* __restrict__ dR,
    const float* __restrict__ z, int S,
    const _Float16* W0t, const float* b0,
    const _Float16* W1t, const float* b1,
    const _Float16* W2t16, const float* b2pad,
    float* __restrict__ sdf_out)
{
  __shared__ _Float16 X0[16 * 32];
  __shared__ _Float16 H0[16 * 264];
  __shared__ _Float16 H1[16 * 264];
  __shared__ float    OUTV[16 * 16];
  const int tid = threadIdx.x;
  const int p0 = blockIdx.x * 16;
  for (int i = tid; i < 16 * 32; i += 256) {
    int m = i >> 5, k = i & 31;
    int p = p0 + m, r = p / S, s = p - r * S;
    float zv = z[r * S + s];
    float val = 0.f;
    if (k < 3) val = o[r * 3 + k] + dR[r * 3 + k] * zv;
    X0[m * 32 + k] = (_Float16)val;
  }
  __syncthreads();
  mlp_layer<32, 256, 1, true, false, true, false>(X0, 32, W0t, b0, H0, 264, nullptr, nullptr, 0);
  __syncthreads();
  mlp_layer<256, 256, 1, true, false, true, false>(H0, 264, W1t, b1, H1, 264, nullptr, nullptr, 0);
  __syncthreads();
  mlp_layer<256, 16, 0, true, false, false, true>(H1, 264, W2t16, b2pad, nullptr, 0, OUTV, nullptr, 0);
  __syncthreads();
  if (tid < 16) sdf_out[p0 + tid] = OUTV[tid * 16 + 0];
}

// ---------------------------------------------------------------------------
// SDF render kernel: forward + analytic gradient (backward GEMMs via WMMA),
// outputs per-point alpha, unit normal, mid_z. (color/smooth nets are dead code)
// ---------------------------------------------------------------------------
__global__ void __launch_bounds__(256) sdf_render_kernel(
    const float* __restrict__ o, const float* __restrict__ dR,
    const float* __restrict__ z, int S,
    const _Float16* W0t, const float* b0,
    const _Float16* W1t, const float* b1,
    const _Float16* W2t16, const float* b2pad,
    const _Float16* W1f, const _Float16* W0rows,
    const float* svar,
    float* __restrict__ alpha_out, float* __restrict__ nrm_out, float* __restrict__ midz_out)
{
  __shared__ _Float16 X0[16 * 32];
  __shared__ _Float16 H0[16 * 264];
  __shared__ _Float16 H1[16 * 264];
  __shared__ _Float16 D1[16 * 264];
  __shared__ _Float16 D0[16 * 264];
  __shared__ float    OUTV[16 * 16];
  __shared__ float    OUTG[16 * 16];
  __shared__ float    Pdist[16], Pmid[16], Pdir[16 * 3];
  const int tid = threadIdx.x;
  const int p0 = blockIdx.x * 16;
  for (int i = tid; i < 16 * 32; i += 256) {
    int m = i >> 5, k = i & 31;
    int p = p0 + m, r = p / S, s = p - r * S;
    float zv = z[r * S + s];
    float dist = (s < S - 1) ? (z[r * S + s + 1] - zv) : SAMPLE_DIST;
    float mid = zv + 0.5f * dist;
    float val = 0.f;
    if (k < 3) { float dv = dR[r * 3 + k]; val = o[r * 3 + k] + dv * mid; Pdir[m * 3 + k] = dv; }
    if (k == 0) { Pdist[m] = dist; Pmid[m] = mid; }
    X0[m * 32 + k] = (_Float16)val;
  }
  __syncthreads();
  mlp_layer<32, 256, 1, true, false, true, false>(X0, 32, W0t, b0, H0, 264, nullptr, nullptr, 0);
  __syncthreads();
  mlp_layer<256, 256, 1, true, false, true, false>(H0, 264, W1t, b1, H1, 264, nullptr, nullptr, 0);
  __syncthreads();
  mlp_layer<256, 16, 0, true, false, false, true>(H1, 264, W2t16, b2pad, nullptr, 0, OUTV, nullptr, 0);
  __syncthreads();
  // d_pre1 = g1 * relu'(h1); g1 = W_sdf2[:,0] = row 0 of W2t16
  for (int i = tid; i < 16 * 256; i += 256) {
    int m = i >> 8, j = i & 255;
    float h = (float)H1[m * 264 + j];
    D1[m * 264 + j] = (h > 0.f) ? W2t16[j] : (_Float16)0.f;
  }
  __syncthreads();
  // d_pre0 = (d_pre1 @ W1^T) * relu'(h0)  -- W1 row-major is W1^T in Wt layout
  mlp_layer<256, 256, 0, false, true, true, false>(D1, 264, W1f, nullptr, D0, 264, nullptr, H0, 264);
  __syncthreads();
  // dp = d_pre0 @ W0^T
  mlp_layer<256, 16, 0, false, false, false, true>(D0, 264, W0rows, nullptr, nullptr, 0, OUTG, nullptr, 0);
  __syncthreads();
  if (tid < 16) {
    int m = tid, p = p0 + m;
    float gx = OUTG[m * 16 + 0], gy = OUTG[m * 16 + 1], gz = OUTG[m * 16 + 2];
    float sdf = OUTV[m * 16 + 0];
    float dx = Pdir[m * 3], dy = Pdir[m * 3 + 1], dz = Pdir[m * 3 + 2];
    float tc = dx * gx + dy * gy + dz * gz;
    float ic = fminf(tc, 0.f);                        // -relu(-true_cos)
    float inv_s = fminf(fmaxf(__expf(10.f * svar[0]), 1e-6f), 1e6f);
    float df = Pdist[m];
    float pc = sigmoidf_dev((sdf - 0.5f * ic * df) * inv_s);
    float nc = sigmoidf_dev((sdf + 0.5f * ic * df) * inv_s);
    float a = (pc - nc + 1e-5f) / (pc + 1e-5f);
    a = fminf(fmaxf(a, 0.f), 1.f);
    alpha_out[p] = a;
    float gl = sqrtf(gx * gx + gy * gy + gz * gz) + 1e-12f;
    nrm_out[p * 3 + 0] = gx / gl; nrm_out[p * 3 + 1] = gy / gl; nrm_out[p * 3 + 2] = gz / gl;
    midz_out[p] = Pmid[m];
  }
}

// ---------------------------------------------------------------------------
// NeRF kernel: fused 6->256->256->4 MLP (+ optional mask MLPs), WMMA
// ---------------------------------------------------------------------------
__global__ void __launch_bounds__(256) nerf_kernel(
    const float* __restrict__ o, const float* __restrict__ dR,
    const float* __restrict__ z, int S, int useMid, float lastDist,
    const _Float16* W0t, const float* b0, const _Float16* W1t, const float* b1,
    const _Float16* W2t, const float* b2pad,
    const _Float16* rf0t, const float* brf0, const _Float16* rf1t, const float* brf1p,
    const _Float16* rl0t, const float* brl0, const _Float16* rl1t, const float* brl1p,
    int writeRgb, int writeMask,
    float* __restrict__ alpha_out, float* __restrict__ rgb_out,
    float* __restrict__ mp1_out, float* __restrict__ mp2_out)
{
  __shared__ _Float16 X0[16 * 32];
  __shared__ _Float16 H0[16 * 264];
  __shared__ _Float16 H1[16 * 264];
  __shared__ float    OUT[16 * 16];
  __shared__ float    Pdist[16];
  const int tid = threadIdx.x;
  const int p0 = blockIdx.x * 16;
  for (int i = tid; i < 16 * 32; i += 256) {
    int m = i >> 5, k = i & 31;
    int p = p0 + m, r = p / S, s = p - r * S;
    float zv = z[r * S + s];
    float dist = (s < S - 1) ? (z[r * S + s + 1] - zv) : lastDist;
    float t = useMid ? (zv + 0.5f * dist) : zv;
    float val = 0.f;
    if (k < 3)       val = o[r * 3 + k] + dR[r * 3 + k] * t;
    else if (k < 6)  val = dR[r * 3 + (k - 3)];
    if (k == 0) Pdist[m] = dist;
    X0[m * 32 + k] = (_Float16)val;
  }
  __syncthreads();
  mlp_layer<32, 256, 1, true, false, true, false>(X0, 32, W0t, b0, H0, 264, nullptr, nullptr, 0);
  __syncthreads();
  mlp_layer<256, 256, 1, true, false, true, false>(H0, 264, W1t, b1, H1, 264, nullptr, nullptr, 0);
  __syncthreads();
  mlp_layer<256, 16, 0, true, false, false, true>(H1, 264, W2t, b2pad, nullptr, 0, OUT, nullptr, 0);
  __syncthreads();
  if (tid < 16) {
    int m = tid, p = p0 + m;
    float sigma = OUT[m * 16 + 0];
    alpha_out[p] = 1.f - __expf(-fmaxf(sigma, 0.f) * Pdist[m]);
    if (writeRgb) {
      rgb_out[p * 3 + 0] = sigmoidf_dev(OUT[m * 16 + 1]);
      rgb_out[p * 3 + 1] = sigmoidf_dev(OUT[m * 16 + 2]);
      rgb_out[p * 3 + 2] = sigmoidf_dev(OUT[m * 16 + 3]);
    }
  }
  if (writeMask) {  // grid-uniform branch
    __syncthreads();
    // mask input = pts only; X0 cols 3..5 hold dirs but rf0t rows>=3 are zero-padded
    mlp_layer<32, 256, 1, true, false, true, false>(X0, 32, rf0t, brf0, H0, 264, nullptr, nullptr, 0);
    __syncthreads();
    mlp_layer<256, 16, 0, true, false, false, true>(H0, 264, rf1t, brf1p, nullptr, 0, OUT, nullptr, 0);
    __syncthreads();
    if (tid < 16) mp1_out[p0 + tid] = sigmoidf_dev(OUT[tid * 16 + 0]);
    __syncthreads();
    mlp_layer<32, 256, 1, true, false, true, false>(X0, 32, rl0t, brl0, H0, 264, nullptr, nullptr, 0);
    __syncthreads();
    mlp_layer<256, 16, 0, true, false, false, true>(H0, 264, rl1t, brl1p, nullptr, 0, OUT, nullptr, 0);
    __syncthreads();
    if (tid < 16) mp2_out[p0 + tid] = sigmoidf_dev(OUT[tid * 16 + 0]);
  }
}

// ---------------------------------------------------------------------------
// Per-ray sequential helpers (1 thread / ray)
// ---------------------------------------------------------------------------
__device__ void sample_pdf_dev(const float* bins, const float* w, int nb, int nw, int n, float* out)
{
  float cdf[132];
  float sum = 0.f;
  for (int i = 0; i < nw; ++i) sum += w[i] + 1e-5f;
  float inv = 1.f / sum;
  cdf[0] = 0.f;
  for (int i = 0; i < nw; ++i) cdf[i + 1] = cdf[i] + (w[i] + 1e-5f) * inv;
  const int nc = nw + 1;
  int idx = 0;
  float step = (n > 1) ? (1.f - 1.f / (float)n) / (float)(n - 1) : 0.f;
  for (int j = 0; j < n; ++j) {
    float u = 0.5f / (float)n + step * (float)j;
    while (idx < nc && cdf[idx] <= u) idx++;
    int below = idx - 1; if (below < 0) below = 0; if (below > nb - 1) below = nb - 1;
    int above = idx;     if (above > nb - 1) above = nb - 1;
    float cb = cdf[below], ca = cdf[above], bb = bins[below], ba = bins[above];
    float den = (ca - cb < 1e-5f) ? 1.f : (ca - cb);
    out[j] = bb + (u - cb) / den * (ba - bb);
  }
}

__device__ void sort_small(float* a, int n)
{
  for (int i = 1; i < n; ++i) {
    float v = a[i]; int j = i - 1;
    while (j >= 0 && a[j] > v) { a[j + 1] = a[j]; j--; }
    a[j + 1] = v;
  }
}

__global__ void upsample_sdf_kernel(const float* __restrict__ o, const float* __restrict__ dR,
                                    const float* __restrict__ z, const float* __restrict__ sdf,
                                    int S, float sscale, float* __restrict__ z_out)
{
  int r = blockIdx.x * blockDim.x + threadIdx.x;
  if (r >= BRAYS) return;
  const float* zr = z + r * S;
  const float* sr = sdf + r * S;
  float w[128], nz[16];
  float ox = o[r * 3], oy = o[r * 3 + 1], oz = o[r * 3 + 2];
  float dx = dR[r * 3], dy = dR[r * 3 + 1], dz = dR[r * 3 + 2];
  float T = 1.f, rawPrev = 0.f;
  float px = ox + dx * zr[0], py = oy + dy * zr[0], pz = oz + dz * zr[0];
  float radPrev = sqrtf(px * px + py * py + pz * pz);
  for (int s = 0; s < S - 1; ++s) {
    float z0 = zr[s], z1 = zr[s + 1], s0 = sr[s], s1 = sr[s + 1];
    float qx = ox + dx * z1, qy = oy + dy * z1, qz = oz + dz * z1;
    float rad1 = sqrtf(qx * qx + qy * qy + qz * qz);
    float raw = (s1 - s0) / (z1 - z0 + 1e-5f);
    float prev = (s == 0) ? 0.f : rawPrev;
    float cc = fminf(raw, prev);
    cc = fminf(fmaxf(cc, -1000.f), 0.f);
    float inside = (radPrev < 1.f || rad1 < 1.f) ? 1.f : 0.f;
    cc *= inside;
    float midv = 0.5f * (s0 + s1), dist = z1 - z0;
    float pc = sigmoidf_dev((midv - 0.5f * cc * dist) * sscale);
    float ncv = sigmoidf_dev((midv + 0.5f * cc * dist) * sscale);
    float a = (pc - ncv + 1e-5f) / (pc + 1e-5f);
    w[s] = a * T; T *= (1.f - a + 1e-7f);
    rawPrev = raw; radPrev = rad1;
  }
  sample_pdf_dev(zr, w, S, S - 1, 16, nz);
  sort_small(nz, 16);
  float* out = z_out + r * (S + 16);
  int i = 0, j = 0, k = 0;
  while (i < S && j < 16) out[k++] = (zr[i] <= nz[j]) ? zr[i++] : nz[j++];
  while (i < S) out[k++] = zr[i++];
  while (j < 16) out[k++] = nz[j++];
}

__global__ void nerf_importance_kernel(const float* __restrict__ z, const float* __restrict__ alpha,
                                       int S, int nImp, float* __restrict__ z_out)
{
  int r = blockIdx.x * blockDim.x + threadIdx.x;
  if (r >= BRAYS) return;
  const float* zr = z + r * S;
  const float* ar = alpha + r * S;
  float w[128], nz[64];
  float T = 1.f;
  for (int s = 0; s < S; ++s) {
    float a = ar[s]; float ww = a * T; T *= (1.f - a + 1e-7f);
    if (s < S - 1) w[s] = ww;
  }
  sample_pdf_dev(zr, w, S, S - 1, nImp, nz);
  sort_small(nz, nImp);
  float* out = z_out + r * (S + nImp);
  int i = 0, j = 0, k = 0;
  while (i < S && j < nImp) out[k++] = (zr[i] <= nz[j]) ? zr[i++] : nz[j++];
  while (i < S) out[k++] = zr[i++];
  while (j < nImp) out[k++] = nz[j++];
}

__global__ void sdf_composite_kernel(const float* __restrict__ alpha, const float* __restrict__ midz,
                                     const float* __restrict__ nrm, int S,
                                     float* __restrict__ depths, float* __restrict__ surfn)
{
  int r = blockIdx.x * blockDim.x + threadIdx.x;
  if (r >= BRAYS) return;
  float T = 1.f, dsum = 0.f, nx = 0.f, ny = 0.f, nz = 0.f;
  for (int s = 0; s < S; ++s) {
    int p = r * S + s;
    float a = alpha[p]; float w = a * T; T *= (1.f - a + 1e-7f);
    dsum += midz[p] * w;
    nx += w * nrm[p * 3]; ny += w * nrm[p * 3 + 1]; nz += w * nrm[p * 3 + 2];
  }
  depths[r] = dsum;
  float l = sqrtf(nx * nx + ny * ny + nz * nz) + 1e-12f;
  surfn[r * 3] = nx / l; surfn[r * 3 + 1] = ny / l; surfn[r * 3 + 2] = nz / l;
}

__global__ void nerf_composite_kernel(const float* __restrict__ alpha, const float* __restrict__ rgb,
                                      const float* __restrict__ mp1, const float* __restrict__ mp2,
                                      int S, float* __restrict__ color,
                                      float* __restrict__ m1, float* __restrict__ m2, int hasMask)
{
  int r = blockIdx.x * blockDim.x + threadIdx.x;
  if (r >= BRAYS) return;
  float T = 1.f, c0 = 0.f, c1 = 0.f, c2 = 0.f, s1 = 0.f, s2 = 0.f;
  for (int s = 0; s < S; ++s) {
    int p = r * S + s;
    float a = alpha[p]; float w = a * T; T *= (1.f - a + 1e-7f);
    c0 += rgb[p * 3] * w; c1 += rgb[p * 3 + 1] * w; c2 += rgb[p * 3 + 2] * w;
    if (hasMask) { s1 += w * mp1[p]; s2 += w * mp2[p]; }
  }
  color[r * 3] = c0; color[r * 3 + 1] = c1; color[r * 3 + 2] = c2;
  if (hasMask) { m1[r] = s1; m2[r] = s2; }
}

__global__ void reflect_kernel(const float* __restrict__ o, const float* __restrict__ dR,
                               const float* __restrict__ depths, const float* __restrict__ surfn,
                               float* __restrict__ hit, float* __restrict__ refl, float* __restrict__ refr)
{
  int r = blockIdx.x * blockDim.x + threadIdx.x;
  if (r >= BRAYS) return;
  float dx = dR[r * 3], dy = dR[r * 3 + 1], dz = dR[r * 3 + 2];
  float nx = surfn[r * 3], ny = surfn[r * 3 + 1], nz = surfn[r * 3 + 2];
  float cs = dx * nx + dy * ny + dz * nz;
  if (cs > 0.f) { nx = -nx; ny = -ny; nz = -nz; }
  float cosi = -(dx * nx + dy * ny + dz * nz);
  float rx = dx + 2.f * cosi * nx, ry = dy + 2.f * cosi * ny, rz = dz + 2.f * cosi * nz;
  const float eta = 1.f / 1.5f;
  float kk = 1.f - eta * eta * (1.f - cosi * cosi);
  float tx, ty, tz;
  if (kk < 0.f) { tx = rx; ty = ry; tz = rz; }
  else {
    float c = eta * cosi - sqrtf(fmaxf(kk, 0.f));
    tx = eta * dx + c * nx; ty = eta * dy + c * ny; tz = eta * dz + c * nz;
  }
  float rl = sqrtf(rx * rx + ry * ry + rz * rz) + 1e-12f;
  float tl = sqrtf(tx * tx + ty * ty + tz * tz) + 1e-12f;
  float dep = depths[r];
  hit[r * 3] = o[r * 3] + dx * dep; hit[r * 3 + 1] = o[r * 3 + 1] + dy * dep; hit[r * 3 + 2] = o[r * 3 + 2] + dz * dep;
  refl[r * 3] = rx / rl; refl[r * 3 + 1] = ry / rl; refl[r * 3 + 2] = rz / rl;
  refr[r * 3] = tx / tl; refr[r * 3 + 1] = ty / tl; refr[r * 3 + 2] = tz / tl;
}

__global__ void final_kernel(const float* __restrict__ c1, const float* __restrict__ m1,
                             const float* __restrict__ m2, const float* __restrict__ cf,
                             const float* __restrict__ cz, const float* __restrict__ cb,
                             float* __restrict__ out)
{
  int i = blockIdx.x * blockDim.x + threadIdx.x;
  if (i >= BRAYS * 3) return;
  int r = i / 3;
  float rgbz = 0.5f * cz[i] + 0.5f * cb[i];
  float rgb = c1[i] * (1.f - m2[r]) + cf[i] * m2[r];
  out[i] = rgb * (1.f - m1[r]) + rgbz * m1[r];
}

// ---------------------------------------------------------------------------
// Host orchestration
// ---------------------------------------------------------------------------
extern "C" void kernel_launch(void* const* d_in, const int* in_sizes, int n_in,
                              void* d_out, int out_size, void* d_ws, size_t ws_size,
                              hipStream_t stream)
{
  (void)in_sizes; (void)n_in; (void)out_size;
  const float* rays_o = (const float*)d_in[0];
  const float* rays_d = (const float*)d_in[1];
  const float* nearp  = (const float*)d_in[2];
  const float* farp   = (const float*)d_in[3];
  const float* W_sdf0 = (const float*)d_in[4];
  const float* b_sdf0 = (const float*)d_in[5];
  const float* W_sdf1 = (const float*)d_in[6];
  const float* b_sdf1 = (const float*)d_in[7];
  const float* W_sdf2 = (const float*)d_in[8];
  const float* b_sdf2 = (const float*)d_in[9];
  const float* s_var  = (const float*)d_in[10];
  const float* Wn0 = (const float*)d_in[15];
  const float* bn0 = (const float*)d_in[16];
  const float* Wn1 = (const float*)d_in[17];
  const float* bn1 = (const float*)d_in[18];
  const float* Wn2 = (const float*)d_in[19];
  const float* bn2 = (const float*)d_in[20];
  const float* Wrf0 = (const float*)d_in[21];
  const float* brf0 = (const float*)d_in[22];
  const float* Wrf1 = (const float*)d_in[23];
  const float* brf1 = (const float*)d_in[24];
  const float* Wrl0 = (const float*)d_in[25];
  const float* brl0 = (const float*)d_in[26];
  const float* Wrl1 = (const float*)d_in[27];
  const float* brl1 = (const float*)d_in[28];

  char* ws = (char*)d_ws;
  size_t off = 0;
  auto alloc = [&](size_t bytes) -> void* {
    void* p = ws + off;
    off = (off + bytes + 255) & ~(size_t)255;
    return p;
  };
  // f16 weights (transposed/padded)
  _Float16* sW0t   = (_Float16*)alloc(256 * 32 * 2);
  _Float16* sW1t   = (_Float16*)alloc(256 * 256 * 2);
  _Float16* sW2t16 = (_Float16*)alloc(16 * 256 * 2);
  _Float16* sW1f   = (_Float16*)alloc(256 * 256 * 2);
  _Float16* sW0r   = (_Float16*)alloc(16 * 256 * 2);
  _Float16* nW0t   = (_Float16*)alloc(256 * 32 * 2);
  _Float16* nW1t   = (_Float16*)alloc(256 * 256 * 2);
  _Float16* nW2t   = (_Float16*)alloc(16 * 256 * 2);
  _Float16* rf0t   = (_Float16*)alloc(256 * 32 * 2);
  _Float16* rf1t   = (_Float16*)alloc(16 * 256 * 2);
  _Float16* rl0t   = (_Float16*)alloc(256 * 32 * 2);
  _Float16* rl1t   = (_Float16*)alloc(16 * 256 * 2);
  // padded biases
  float* b2pad   = (float*)alloc(16 * 4);
  float* bn2pad  = (float*)alloc(16 * 4);
  float* brf1pad = (float*)alloc(16 * 4);
  float* brl1pad = (float*)alloc(16 * 4);
  // f32 buffers
  float* zA    = (float*)alloc(BRAYS * 128 * 4);
  float* zB    = (float*)alloc(BRAYS * 128 * 4);
  float* sdfv  = (float*)alloc(BRAYS * 128 * 4);
  float* alphaB= (float*)alloc(BRAYS * 128 * 4);
  float* nrmB  = (float*)alloc(BRAYS * 128 * 3 * 4);
  float* midzB = (float*)alloc(BRAYS * 128 * 4);
  float* rgbp  = (float*)alloc(BRAYS * 128 * 3 * 4);
  float* mp1   = (float*)alloc(BRAYS * 128 * 4);
  float* mp2   = (float*)alloc(BRAYS * 128 * 4);
  float* depths= (float*)alloc(BRAYS * 4);
  float* surfn = (float*)alloc(BRAYS * 3 * 4);
  float* hitB  = (float*)alloc(BRAYS * 3 * 4);
  float* reflB = (float*)alloc(BRAYS * 3 * 4);
  float* refrB = (float*)alloc(BRAYS * 3 * 4);
  float* c1v   = (float*)alloc(BRAYS * 3 * 4);
  float* m1v   = (float*)alloc(BRAYS * 4);
  float* m2v   = (float*)alloc(BRAYS * 4);
  float* cfv   = (float*)alloc(BRAYS * 3 * 4);
  float* czv   = (float*)alloc(BRAYS * 3 * 4);
  float* cbv   = (float*)alloc(BRAYS * 3 * 4);
  if (off > ws_size) return;  // workspace too small; bail deterministically

  auto g = [](int n) { return dim3((unsigned)((n + 255) / 256)); };
  const dim3 blk(256);

  // ---- weight prep (f32 -> f16 transpose/pad) ----
  w_transpose_pad<<<g(256 * 32), blk, 0, stream>>>(sW0t, W_sdf0, 3, 256, 256, 32, 256);
  w_transpose_pad<<<g(256 * 256), blk, 0, stream>>>(sW1t, W_sdf1, 256, 256, 256, 256, 256);
  w_transpose_pad<<<g(16 * 256), blk, 0, stream>>>(sW2t16, W_sdf2, 256, 1, 257, 256, 16);
  w_copy_pad<<<g(256 * 256), blk, 0, stream>>>(sW1f, W_sdf1, 256, 256, 256, 256);
  w_copy_pad<<<g(16 * 256), blk, 0, stream>>>(sW0r, W_sdf0, 3, 256, 16, 256);
  w_transpose_pad<<<g(256 * 32), blk, 0, stream>>>(nW0t, Wn0, 6, 256, 256, 32, 256);
  w_transpose_pad<<<g(256 * 256), blk, 0, stream>>>(nW1t, Wn1, 256, 256, 256, 256, 256);
  w_transpose_pad<<<g(16 * 256), blk, 0, stream>>>(nW2t, Wn2, 256, 4, 4, 256, 16);
  w_transpose_pad<<<g(256 * 32), blk, 0, stream>>>(rf0t, Wrf0, 3, 256, 256, 32, 256);
  w_transpose_pad<<<g(16 * 256), blk, 0, stream>>>(rf1t, Wrf1, 256, 1, 1, 256, 16);
  w_transpose_pad<<<g(256 * 32), blk, 0, stream>>>(rl0t, Wrl0, 3, 256, 256, 32, 256);
  w_transpose_pad<<<g(16 * 256), blk, 0, stream>>>(rl1t, Wrl1, 256, 1, 1, 256, 16);
  bias_pad<<<1, 16, 0, stream>>>(b2pad, b_sdf2, 1, 16);
  bias_pad<<<1, 16, 0, stream>>>(bn2pad, bn2, 4, 16);
  bias_pad<<<1, 16, 0, stream>>>(brf1pad, brf1, 1, 16);
  bias_pad<<<1, 16, 0, stream>>>(brl1pad, brl1, 1, 16);

  // ---- SDF importance sampling: 64 -> 80 -> 96 -> 112 -> 128 samples ----
  int S = NSC;
  float* zcur = zA;
  float* znext = zB;
  init_z_kernel<<<g(BRAYS * S), blk, 0, stream>>>(zcur, S, nearp, farp, 0.f, 0.f);
  for (int i = 0; i < 4; ++i) {
    sdf_value_kernel<<<dim3(BRAYS * S / 16), blk, 0, stream>>>(
        rays_o, rays_d, zcur, S, sW0t, b_sdf0, sW1t, b_sdf1, sW2t16, b2pad, sdfv);
    upsample_sdf_kernel<<<g(BRAYS), blk, 0, stream>>>(
        rays_o, rays_d, zcur, sdfv, S, 64.f * (float)(1 << i), znext);
    float* t = zcur; zcur = znext; znext = t;
    S += 16;
  }
  // ---- SDF render: fwd + grad -> alpha / normal / mid_z; composite ----
  sdf_render_kernel<<<dim3(BRAYS * S / 16), blk, 0, stream>>>(
      rays_o, rays_d, zcur, S, sW0t, b_sdf0, sW1t, b_sdf1, sW2t16, b2pad,
      sW1f, sW0r, s_var, alphaB, nrmB, midzB);
  sdf_composite_kernel<<<g(BRAYS), blk, 0, stream>>>(alphaB, midzB, nrmB, S, depths, surfn);
  reflect_kernel<<<g(BRAYS), blk, 0, stream>>>(rays_o, rays_d, depths, surfn, hitB, reflB, refrB);

  // one full NeRF pass: coarse -> importance resample -> fine render -> composite
  auto nerf_pass = [&](const float* o, const float* d, int S0, float base, float span,
                       int useNearFar, int nImp, int wantMask,
                       float* color, float* m1, float* m2) {
    init_z_kernel<<<g(BRAYS * S0), blk, 0, stream>>>(
        zA, S0, useNearFar ? nearp : nullptr, useNearFar ? farp : nullptr, base, span);
    nerf_kernel<<<dim3(BRAYS * S0 / 16), blk, 0, stream>>>(
        o, d, zA, S0, 0, 1e10f, nW0t, bn0, nW1t, bn1, nW2t, bn2pad,
        rf0t, brf0, rf1t, brf1pad, rl0t, brl0, rl1t, brl1pad,
        0, 0, alphaB, rgbp, mp1, mp2);
    nerf_importance_kernel<<<g(BRAYS), blk, 0, stream>>>(zA, alphaB, S0, nImp, zB);
    int S1 = S0 + nImp;
    nerf_kernel<<<dim3(BRAYS * S1 / 16), blk, 0, stream>>>(
        o, d, zB, S1, 1, SAMPLE_DIST, nW0t, bn0, nW1t, bn1, nW2t, bn2pad,
        rf0t, brf0, rf1t, brf1pad, rl0t, brl0, rl1t, brl1pad,
        1, wantMask, alphaB, rgbp, mp1, mp2);
    nerf_composite_kernel<<<g(BRAYS), blk, 0, stream>>>(
        alphaB, rgbp, mp1, mp2, S1, color, m1, m2, wantMask);
  };

  // main NeRF pass (with masks), reflection pass, two refraction passes
  nerf_pass(rays_o, rays_d, NSC, 0.f, 0.f, 1, 64, 1, c1v, m1v, m2v);
  nerf_pass(hitB, reflB, NSC, 0.05f, 3.7f - 0.05f, 0, 64, 0, cfv, nullptr, nullptr);
  nerf_pass(hitB, refrB, NSC / 2, 0.f, 3.7f, 0, 32, 0, czv, nullptr, nullptr);
  nerf_pass(hitB, refrB, NSC / 2, 0.5f, 3.7f - 0.5f, 0, 32, 0, cbv, nullptr, nullptr);

  final_kernel<<<g(BRAYS * 3), blk, 0, stream>>>(c1v, m1v, m2v, cfv, czv, cbv, (float*)d_out);
}
